// Attention_26079041421348
// MI455X (gfx1250) — compile-verified, matrix operand order checked by reading, not run on hardware
//
#include <hip/hip_runtime.h>
#include <hip/hip_bf16.h>

typedef __attribute__((ext_vector_type(16))) __bf16 v16bf;
typedef __attribute__((ext_vector_type(8)))  float  v8f;

__device__ __forceinline__ unsigned short f2bf(float f) {
    // round-to-nearest-even fp32 -> bf16
    unsigned int u = __float_as_uint(f);
    u += 0x7FFFu + ((u >> 16) & 1u);
    return (unsigned short)(u >> 16);
}

// CDNA5 has a hardware transcendental V_TANH_F32 (TRANS32 class, co-executes
// with WMMA). Prefer it; fall back to a branch-free native-exp formulation.
__device__ __forceinline__ float fast_tanh(float x) {
#if __has_builtin(__builtin_amdgcn_tanhf)
    return __builtin_amdgcn_tanhf(x);
#else
    return 1.0f - 2.0f / (__expf(2.0f * x) + 1.0f);
#endif
}

union BfVec {
    v16bf v;
    unsigned short u[16];
    uint4 q[2];
};

// ---------------------------------------------------------------------------
// Kernel A: scores[r] = tanh(X[r,:] @ W1 + b1) @ W2   for r in [0, B*T)
// One wave per 16-row tile. K = 256 (8 WMMA steps of 32), N = 128 (8 tiles).
// ---------------------------------------------------------------------------
__global__ void __launch_bounds__(256) score_kernel(
    const float* __restrict__ X,    // (B*T, 256) row-major
    const float* __restrict__ W1,   // (256, 128) row-major
    const float* __restrict__ b1,   // (128)
    const float* __restrict__ W2,   // (128)
    float* __restrict__ scores)     // (B*T)
{
    // W1 transposed -> bf16 in LDS: w1t[n*256 + k] = bf16(W1[k*128 + n])
    __shared__ unsigned short w1t[128 * 256];   // 64 KB

    const int tid = threadIdx.x;
    for (int idx = tid; idx < 128 * 256; idx += 256) {
        const int n = idx >> 8;
        const int k = idx & 255;
        w1t[idx] = f2bf(W1[k * 128 + n]);
    }
    __syncthreads();

    const int lane  = tid & 31;
    const int wave  = tid >> 5;
    const int tile  = blockIdx.x * 8 + wave;        // 16-row tile index
    const int row   = (tile << 4) + (lane & 15);    // this lane's A row
    const int khalf = (lane >> 4) ? 8 : 0;          // ISA A/B lane split
    const float* __restrict__ xrow = X + (size_t)row * 256;

    float part[8] = {0.f, 0.f, 0.f, 0.f, 0.f, 0.f, 0.f, 0.f};

    for (int nt = 0; nt < 8; ++nt) {
        const int col = (nt << 4) + (lane & 15);    // this lane's C column
        v8f c = {};                                 // f32 accumulator (8 VGPRs)
        #pragma unroll
        for (int kt = 0; kt < 8; ++kt) {
            const int kb = (kt << 5) + khalf;       // base K for this lane

            // A fragment: row 'row', K = kb..kb+7 and kb+16..kb+23 (fp32 -> bf16)
            const float4 f0 = *(const float4*)(xrow + kb);
            const float4 f1 = *(const float4*)(xrow + kb + 4);
            const float4 f2 = *(const float4*)(xrow + kb + 16);
            const float4 f3 = *(const float4*)(xrow + kb + 20);
            BfVec a;
            a.u[0]  = f2bf(f0.x); a.u[1]  = f2bf(f0.y);
            a.u[2]  = f2bf(f0.z); a.u[3]  = f2bf(f0.w);
            a.u[4]  = f2bf(f1.x); a.u[5]  = f2bf(f1.y);
            a.u[6]  = f2bf(f1.z); a.u[7]  = f2bf(f1.w);
            a.u[8]  = f2bf(f2.x); a.u[9]  = f2bf(f2.y);
            a.u[10] = f2bf(f2.z); a.u[11] = f2bf(f2.w);
            a.u[12] = f2bf(f3.x); a.u[13] = f2bf(f3.y);
            a.u[14] = f2bf(f3.z); a.u[15] = f2bf(f3.w);

            // B fragment: column 'col', same K pattern, contiguous in LDS
            BfVec b;
            b.q[0] = *(const uint4*)&w1t[col * 256 + kb];
            b.q[1] = *(const uint4*)&w1t[col * 256 + kb + 16];

            c = __builtin_amdgcn_wmma_f32_16x16x32_bf16(
                    false, a.v, false, b.v, (short)0, c, false, false);
        }
        // fuse bias + tanh + dot with W2 (b2 cancels in the softmax)
        const float bias = b1[col];
        const float w2v  = W2[col];
        #pragma unroll
        for (int i = 0; i < 8; ++i)
            part[i] += fast_tanh(c[i] + bias) * w2v;
    }

    // reduce the 16 column-lanes of each half-wave (rows 0-7 / 8-15)
    #pragma unroll
    for (int m = 8; m >= 1; m >>= 1) {
        #pragma unroll
        for (int i = 0; i < 8; ++i)
            part[i] += __shfl_xor(part[i], m, 32);
    }

    if ((lane & 15) == 0) {
        const int rbase = (tile << 4) + ((lane >> 4) << 3);
        #pragma unroll
        for (int i = 0; i < 8; ++i)
            scores[rbase + i] = part[i];
    }
}

// ---------------------------------------------------------------------------
// Kernel B: attention = softmax(scores, axis=T) per batch. One block per b.
// ---------------------------------------------------------------------------
__global__ void __launch_bounds__(256) softmax_kernel(
    const float* __restrict__ scores,   // (B, 2048)
    float* __restrict__ att)            // (B, 2048)
{
    __shared__ float red[8];
    const int b   = blockIdx.x;
    const int tid = threadIdx.x;
    const float* __restrict__ s = scores + (size_t)b * 2048;

    float v[8];
    float mx = -3.402823466e38f;
    #pragma unroll
    for (int j = 0; j < 8; ++j) {
        v[j] = s[tid + j * 256];
        mx = fmaxf(mx, v[j]);
    }
    #pragma unroll
    for (int m = 16; m >= 1; m >>= 1)
        mx = fmaxf(mx, __shfl_xor(mx, m, 32));
    if ((tid & 31) == 0) red[tid >> 5] = mx;
    __syncthreads();
    float bmax = red[0];
    #pragma unroll
    for (int i = 1; i < 8; ++i) bmax = fmaxf(bmax, red[i]);
    __syncthreads();

    float sum = 0.f;
    #pragma unroll
    for (int j = 0; j < 8; ++j) {
        v[j] = __expf(v[j] - bmax);
        sum += v[j];
    }
    #pragma unroll
    for (int m = 16; m >= 1; m >>= 1)
        sum += __shfl_xor(sum, m, 32);
    if ((tid & 31) == 0) red[tid >> 5] = sum;
    __syncthreads();
    float tot = 0.f;
    #pragma unroll
    for (int i = 0; i < 8; ++i) tot += red[i];

    const float inv = 1.0f / tot;
    #pragma unroll
    for (int j = 0; j < 8; ++j)
        att[(size_t)b * 2048 + tid + j * 256] = v[j] * inv;
}

// ---------------------------------------------------------------------------
// Kernel C: context[b, h] = sum_t att[b,t] * X[b,t,h]. One block per b,
// thread h owns channel h -> fully coalesced 1KB-row streaming.
// ---------------------------------------------------------------------------
__global__ void __launch_bounds__(256) context_kernel(
    const float* __restrict__ X,    // (B, 2048, 256)
    const float* __restrict__ att,  // (B, 2048)
    float* __restrict__ ctx)        // (B, 256)
{
    const int b = blockIdx.x;
    const int h = threadIdx.x;
    const float* __restrict__ xb = X   + (size_t)b * 2048 * 256;
    const float* __restrict__ wb = att + (size_t)b * 2048;

    float a0 = 0.f, a1 = 0.f, a2 = 0.f, a3 = 0.f;
    for (int t = 0; t < 2048; t += 4) {
        a0 += wb[t + 0] * xb[(size_t)(t + 0) * 256 + h];
        a1 += wb[t + 1] * xb[(size_t)(t + 1) * 256 + h];
        a2 += wb[t + 2] * xb[(size_t)(t + 2) * 256 + h];
        a3 += wb[t + 3] * xb[(size_t)(t + 3) * 256 + h];
    }
    ctx[(size_t)b * 256 + h] = (a0 + a1) + (a2 + a3);
}

// ---------------------------------------------------------------------------
extern "C" void kernel_launch(void* const* d_in, const int* in_sizes, int n_in,
                              void* d_out, int out_size, void* d_ws, size_t ws_size,
                              hipStream_t stream) {
    (void)in_sizes; (void)n_in; (void)out_size; (void)ws_size;

    const float* X  = (const float*)d_in[0];   // (128, 2048, 256)
    const float* W1 = (const float*)d_in[1];   // (256, 128)
    const float* b1 = (const float*)d_in[2];   // (128)
    const float* W2 = (const float*)d_in[3];   // (128, 1)
    // d_in[4] = b2: constant shift, cancels in softmax -> unused

    float* ctx    = (float*)d_out;                    // (128, 256)
    float* att    = (float*)d_out + 128 * 256;        // (128, 2048)
    float* scores = (float*)d_ws;                     // (128*2048) scratch

    // 262144 rows / 16 rows-per-wave / 8 waves-per-block = 2048 blocks exactly
    score_kernel  <<<2048, 256, 0, stream>>>(X, W1, b1, W2, scores);
    softmax_kernel<<< 128, 256, 0, stream>>>(scores, att);
    context_kernel<<< 128, 256, 0, stream>>>(X, att, ctx);
}
